// TripletBatchHardLoss_3324304687814
// MI455X (gfx1250) — matrix operand order repaired and checked by reading, not compiled, not run on hardware
//
#include <hip/hip_runtime.h>

// ---------------------------------------------------------------------------
// TripletBatchHardLoss on MI455X (gfx1250, wave32)
//
// Fused bf16-split (hi+lo) WMMA Gram matrix + on-the-fly hard pos/neg row
// reductions. Compute-bound (34 GFLOP vs 8MB data) -> target
// v_wmma_f32_16x16x32_bf16 throughput; B tiles double-buffered in LDS via
// the CDNA5 async-to-LDS engine (ASYNCcnt) when available.
// ---------------------------------------------------------------------------

constexpr int   kN      = 8192;
constexpr int   kD      = 256;
constexpr float kMargin = 1.0f;
constexpr float kEps    = 1e-12f;

typedef __attribute__((ext_vector_type(16))) __bf16 v16bf;
typedef __attribute__((ext_vector_type(8)))  __bf16 v8bf;
typedef __attribute__((ext_vector_type(8)))  unsigned short v8us;
typedef __attribute__((ext_vector_type(8)))  float  v8f;
typedef __attribute__((ext_vector_type(4)))  int    v4i;

// --------------------------- async-to-LDS support --------------------------
#if __has_builtin(__builtin_amdgcn_global_load_async_to_lds_b128) && \
    __has_builtin(__builtin_amdgcn_s_wait_asynccnt)
#define TRIP_ASYNC 1
#else
#define TRIP_ASYNC 0
#endif

// Builtin signature (from hipcc diagnostic): param0 = int4 ptr in AS(1)
// (printed "__device__"), param1 = int4 ptr in AS(3), then imm offset, cpol.
typedef __attribute__((address_space(1))) v4i GB128;
typedef __attribute__((address_space(3))) v4i LB128;

__device__ __forceinline__ void copy16_to_lds(const unsigned short* g,
                                              unsigned short* l) {
#if TRIP_ASYNC
  // global_load_async_to_lds_b128: 16B/lane, tracked by ASYNCcnt, no VGPR data.
  __builtin_amdgcn_global_load_async_to_lds_b128(
      (GB128*)(unsigned long long)g, (LB128*)l, 0, 0);
#else
  *(v8us*)l = *(const v8us*)g;
#endif
}

__device__ __forceinline__ void wait_async_done() {
#if TRIP_ASYNC
  __builtin_amdgcn_s_wait_asynccnt(0);
#endif
}

// ---------------------------------------------------------------------------
// Kernel 1: split f32 embeddings into bf16 hi/lo planes, compute row sumsq,
// init per-row reduction slots.
// ---------------------------------------------------------------------------
__global__ __launch_bounds__(256)
void prep_kernel(const float* __restrict__ emb,
                 unsigned short* __restrict__ hi,
                 unsigned short* __restrict__ lo,
                 float* __restrict__ sq,
                 unsigned* __restrict__ rmax, unsigned* __restrict__ rmin,
                 unsigned* __restrict__ vneg, unsigned* __restrict__ vpos) {
  __shared__ float red[256];
  const int row = blockIdx.x;
  const int t   = threadIdx.x;
  const float x = emb[(size_t)row * kD + t];

  // hi = truncate-to-bf16(x); lo = bf16(x - hi): hi+lo ~ x to ~2^-24.
  const unsigned xb = __float_as_uint(x);
  const unsigned hb = xb & 0xFFFF0000u;
  const float    hf = __uint_as_float(hb);
  const float    lf = x - hf;
  hi[(size_t)row * kD + t] = (unsigned short)(xb >> 16);
  lo[(size_t)row * kD + t] = (unsigned short)(__float_as_uint(lf) >> 16);

  red[t] = x * x;
  __syncthreads();
  for (int s = 128; s > 0; s >>= 1) {
    if (t < s) red[t] += red[t + s];
    __syncthreads();
  }
  if (t == 0) {
    sq[row] = red[0];
    // pdist >= 0 -> IEEE bit pattern is order-preserving as uint:
    rmax[row] = 0u;            // running max        (0.0f)
    vpos[row] = 0u;            // running max (pos)  (0.0f)
    rmin[row] = 0x7F800000u;   // running min        (+inf)
    vneg[row] = 0x7F800000u;   // running min (neg)  (+inf)
  }
}

// ---------------------------------------------------------------------------
// Kernel 2: fused GEMM + masked row reductions.
// Block = 4 waves (128 thr). Wave w owns rows [bx*64 + 16w, +16).
// Block sweeps a 512-column chunk in 32-column stages; each stage's B tile
// (hi+lo bf16, 32KB) is staged into a double-buffered LDS slot while the
// previous stage's 48 WMMAs per wave execute. A stays in VGPRs (col-invariant).
// ---------------------------------------------------------------------------
constexpr int kChunk     = 512;
constexpr int kStageCols = 32;
constexpr int kStages    = kChunk / kStageCols;  // 16

__global__ __launch_bounds__(128)
void tile_kernel(const unsigned short* __restrict__ ehi,
                 const unsigned short* __restrict__ elo,
                 const float* __restrict__ sq,
                 const int* __restrict__ labels,
                 unsigned* __restrict__ rmax, unsigned* __restrict__ rmin,
                 unsigned* __restrict__ vneg, unsigned* __restrict__ vpos) {
  __shared__ unsigned short bufh[2][kStageCols * kD];  // 2 x 16KB hi plane
  __shared__ unsigned short bufl[2][kStageCols * kD];  // 2 x 16KB lo plane

  const int tid  = threadIdx.x;
  const int wave = tid >> 5;
  const int lane = tid & 31;
  const int half = lane >> 4;   // selects K sub-block / C row group
  const int lr   = lane & 15;

  const int rowBase = blockIdx.x * 64 + wave * 16;
  const int colBase = blockIdx.y * kChunk;

  // Per-lane C-row metadata (C layout: VGPR j -> row myRow0 + j, N = lane%16)
  const int myRow0 = rowBase + half * 8;
  float sqr[8];
  int   labr[8];
#pragma unroll
  for (int j = 0; j < 8; ++j) {
    sqr[j]  = sq[myRow0 + j];
    labr[j] = labels[myRow0 + j];
  }

  // Running per-(row j, lane-col) partial reductions.
  float pmax[8], pmin[8], pneg[8], ppos[8];
#pragma unroll
  for (int j = 0; j < 8; ++j) {
    pmax[j] = 0.0f;               ppos[j] = 0.0f;
    pmin[j] = __builtin_inff();   pneg[j] = __builtin_inff();
  }

  const int arow = rowBase + lr;  // A-matrix row this lane supplies

  // Stage issuer: 32 cols x 256 K x 2 planes = 2048 x 16B segments,
  // 128 threads x 8 iterations x 2 planes.
  auto issue_stage = [&](int s) {
    const int b    = s & 1;
    const int colS = colBase + s * kStageCols;
#pragma unroll
    for (int i = 0; i < 8; ++i) {
      const int seg = tid + i * 128;   // 0..1023 per plane
      const int col = seg >> 5;        // 32 x 16B segments per column
      const int ko  = (seg & 31) * 8;
      copy16_to_lds(&ehi[(size_t)(colS + col) * kD + ko], &bufh[b][col * kD + ko]);
      copy16_to_lds(&elo[(size_t)(colS + col) * kD + ko], &bufl[b][col * kD + ko]);
    }
  };

  issue_stage(0);

  for (int s = 0; s < kStages; ++s) {
    wait_async_done();   // my async copies for stage s have landed in LDS
    __syncthreads();     // everyone's copies landed; buf[(s+1)&1] free to fill
    if (s + 1 < kStages) issue_stage(s + 1);  // overlap with compute below

    const int b = s & 1;
#pragma unroll
    for (int t = 0; t < 2; ++t) {      // two 16x16 col tiles per stage
      const int colT = colBase + s * kStageCols + t * 16;
      const int cl   = t * 16 + lr;    // column within stage buffer

      v8f acc = {};
#pragma unroll
      for (int ks = 0; ks < 8; ++ks) {
        const int kk = ks * 32;
        // A tile 16x32 bf16 (v0-3 = K{kk+half*16..+7},
        //                    v4-7 = K{kk+half*16+8..+15})
        v8bf a0h = *(const v8bf*)(&ehi[(size_t)arow * kD + kk + half * 16]);
        v8bf a1h = *(const v8bf*)(&ehi[(size_t)arow * kD + kk + half * 16 + 8]);
        v8bf a0l = *(const v8bf*)(&elo[(size_t)arow * kD + kk + half * 16]);
        v8bf a1l = *(const v8bf*)(&elo[(size_t)arow * kD + kk + half * 16 + 8]);
        v16bf ah = __builtin_shufflevector(a0h, a1h, 0,1,2,3,4,5,6,7,8,9,10,11,12,13,14,15);
        v16bf al = __builtin_shufflevector(a0l, a1l, 0,1,2,3,4,5,6,7,8,9,10,11,12,13,14,15);
        // B tile 32x16 bf16 from LDS (lanes 0-15: K=kk..kk+15, 16-31: +16)
        v16bf bh = *(const v16bf*)(&bufh[b][cl * kD + kk + half * 16]);
        v16bf bl = *(const v16bf*)(&bufl[b][cl * kD + kk + half * 16]);

        // dot += hi*hi + hi*lo + lo*hi   (lo*lo ~ 2^-32, dropped)
        acc = __builtin_amdgcn_wmma_f32_16x16x32_bf16(false, ah, false, bh,
                                                      (short)0, acc, false, false);
        acc = __builtin_amdgcn_wmma_f32_16x16x32_bf16(false, ah, false, bl,
                                                      (short)0, acc, false, false);
        acc = __builtin_amdgcn_wmma_f32_16x16x32_bf16(false, al, false, bh,
                                                      (short)0, acc, false, false);
      }

      // Epilogue: pdist + masked reductions for this 16x16 tile.
      const int   col  = colT + lr;
      const float sqc  = sq[col];
      const int   labc = labels[col];
#pragma unroll
      for (int j = 0; j < 8; ++j) {
        const int   row  = myRow0 + j;
        const float d2   = sqr[j] + sqc - 2.0f * acc[j];
        const float pd   = sqrtf(fmaxf(d2, kEps));  // == sqrt(max(max(d2,0),eps))
        const bool  same = (labr[j] == labc);
        pmax[j] = fmaxf(pmax[j], pd);
        pmin[j] = fminf(pmin[j], pd);
        pneg[j] = same ? pneg[j] : fminf(pneg[j], pd);
        ppos[j] = (same && (row != col)) ? fmaxf(ppos[j], pd) : ppos[j];
      }
    }
  }

  // Reduce partials across the 16 column-lanes of each half, then publish
  // per-row results with order-independent uint min/max atomics.
#pragma unroll
  for (int j = 0; j < 8; ++j) {
    float a = pmax[j], bb = pmin[j], c = pneg[j], d = ppos[j];
    for (int m = 1; m < 16; m <<= 1) {
      a  = fmaxf(a,  __shfl_xor(a,  m, 16));
      bb = fminf(bb, __shfl_xor(bb, m, 16));
      c  = fminf(c,  __shfl_xor(c,  m, 16));
      d  = fmaxf(d,  __shfl_xor(d,  m, 16));
    }
    if (lr == 0) {
      const int row = myRow0 + j;
      atomicMax(&rmax[row], __float_as_uint(a));
      atomicMin(&rmin[row], __float_as_uint(bb));
      atomicMin(&vneg[row], __float_as_uint(c));
      atomicMax(&vpos[row], __float_as_uint(d));
    }
  }
}

// ---------------------------------------------------------------------------
// Kernel 3: per-row loss + deterministic block partial sums.
// hard_neg = min(min_neg, row_max); hard_pos = max(max_pos, row_min)
// (algebraically identical to the reference's masked min/max trick).
// ---------------------------------------------------------------------------
__global__ __launch_bounds__(256)
void loss_kernel(const unsigned* __restrict__ rmax, const unsigned* __restrict__ rmin,
                 const unsigned* __restrict__ vneg, const unsigned* __restrict__ vpos,
                 float* __restrict__ bsum) {
  __shared__ float red[256];
  const int i  = blockIdx.x * 256 + threadIdx.x;
  const float hn = fminf(__uint_as_float(vneg[i]), __uint_as_float(rmax[i]));
  const float hp = fmaxf(__uint_as_float(vpos[i]), __uint_as_float(rmin[i]));
  red[threadIdx.x] = fmaxf(hp - hn + kMargin, 0.0f);
  __syncthreads();
  for (int s = 128; s > 0; s >>= 1) {
    if (threadIdx.x < s) red[threadIdx.x] += red[threadIdx.x + s];
    __syncthreads();
  }
  if (threadIdx.x == 0) bsum[blockIdx.x] = red[0];
}

__global__ void final_kernel(const float* __restrict__ bsum, float* __restrict__ out) {
  float s = 0.0f;
  for (int b = 0; b < kN / 256; ++b) s += bsum[b];
  if (threadIdx.x == 0) out[0] = s * (1.0f / (float)kN);
}

// ---------------------------------------------------------------------------
extern "C" void kernel_launch(void* const* d_in, const int* in_sizes, int n_in,
                              void* d_out, int out_size, void* d_ws, size_t ws_size,
                              hipStream_t stream) {
  (void)in_sizes; (void)n_in; (void)out_size; (void)ws_size;
  const int*   labels = (const int*)d_in[0];   // y_true
  const float* emb    = (const float*)d_in[1]; // y_pred

  char* ws = (char*)d_ws;
  const size_t planeB = (size_t)kN * kD * sizeof(unsigned short);  // 4 MiB
  unsigned short* hi  = (unsigned short*)(ws);
  unsigned short* lo  = (unsigned short*)(ws + planeB);
  float*    sq   = (float*)   (ws + 2 * planeB);
  unsigned* rmax = (unsigned*)(ws + 2 * planeB + 1 * (size_t)kN * 4);
  unsigned* rmin = (unsigned*)(ws + 2 * planeB + 2 * (size_t)kN * 4);
  unsigned* vneg = (unsigned*)(ws + 2 * planeB + 3 * (size_t)kN * 4);
  unsigned* vpos = (unsigned*)(ws + 2 * planeB + 4 * (size_t)kN * 4);
  float*    bsum = (float*)   (ws + 2 * planeB + 5 * (size_t)kN * 4);
  float*    out  = (float*)d_out;

  prep_kernel<<<kN, 256, 0, stream>>>(emb, hi, lo, sq, rmax, rmin, vneg, vpos);

  dim3 grid(kN / 64, kN / kChunk);  // 128 x 16 blocks, 4 waves each
  tile_kernel<<<grid, 128, 0, stream>>>(hi, lo, sq, labels, rmax, rmin, vneg, vpos);

  loss_kernel<<<kN / 256, 256, 0, stream>>>(rmax, rmin, vneg, vpos, bsum);
  final_kernel<<<1, 32, 0, stream>>>(bsum, out);
}
